// GCNEncoder_3968549782293
// MI455X (gfx1250) — matrix operand order repaired and checked by reading, not compiled, not run on hardware
//
#include <hip/hip_runtime.h>

// ---------------------------------------------------------------------------
// Types for WMMA bf16 path (gfx1250 / CDNA5, wave32)
// ---------------------------------------------------------------------------
typedef __attribute__((ext_vector_type(16))) __bf16 v16bf;
typedef __attribute__((ext_vector_type(8)))  __bf16 v8bf;
typedef __attribute__((ext_vector_type(8)))  float  v8f;

union FragAB { v16bf v; v8bf h[2]; };

__device__ __forceinline__ unsigned short f2bf(float f) {
    unsigned int u = __float_as_uint(f);
    u += 0x7fffu + ((u >> 16) & 1u);          // round-to-nearest-even
    return (unsigned short)(u >> 16);
}
__device__ __forceinline__ float bf2f(unsigned short h) {
    return __uint_as_float(((unsigned int)h) << 16);
}

// Async copy of one 32-byte chunk: global -> LDS, bypassing VGPRs.
// LDS address taken from the VDST operand, global address from VADDR;
//      the instruction offset applies to both sides.
// Tracked with ASYNCcnt (in-order completion for loads).
#define ASYNC_CP32(ldsoff, gptr)                                              \
    asm volatile("global_load_async_to_lds_b128 %0, %1, off\n\t"              \
                 "global_load_async_to_lds_b128 %0, %1, off offset:16"        \
                 :: "v"(ldsoff), "v"((unsigned long long)(size_t)(gptr))      \
                 : "memory")

// ---------------------------------------------------------------------------
// Generic bf16 WMMA GEMM:  out = A[M x K](bf16, rowmajor) * Wt[N x K]^T + bias
//   mode 0: outF[row*N+col]  = acc + bias        (f32 store)
//   mode 1: outH[row*N+col]  = bf16(acc + bias)  (bf16 store)
//   mode 2: outF[row*N+col] += acc + bias        (residual accumulate)
// Block: 256 threads = 8 waves; block tile 128x128, BK=32.
// Wave grid 4(M) x 2(N); wave tile 32x64 -> 2x4 fragments of 16x16.
// Double-buffered LDS fed by async global->LDS copies (ASYNCcnt pipeline).
// ---------------------------------------------------------------------------
#define BM 128
#define BN 128
#define BK 32
#define LDSP 40   // padded LDS row stride in bf16 elems (80B = 5*16B, keeps 16B align)

__global__ __launch_bounds__(256, 2) void gemm_bf16_wmma(
    const unsigned short* __restrict__ A,    // [M x K] bf16
    const unsigned short* __restrict__ Bt,   // [N x K] bf16 (pre-transposed weights)
    const float* __restrict__ bias,          // [N]
    float* __restrict__ outF,
    unsigned short* __restrict__ outH,
    int Nn, int K, int mode)
{
    __shared__ unsigned short sA[2][BM * LDSP];
    __shared__ unsigned short sB[2][BN * LDSP];

    const int tid  = threadIdx.x;
    const int lane = tid & 31;
    const int wave = tid >> 5;
    const int wm = wave >> 1;          // 0..3 (M)
    const int wn = wave & 1;           // 0..1 (N)
    const int l16  = lane & 15;
    const int half = lane >> 4;

    const size_t bm = (size_t)blockIdx.x * BM;
    const size_t bn = (size_t)blockIdx.y * BN;

    // each thread stages 16 bf16 (32B) of A and of B per k-tile
    const int ldr = tid >> 1;          // 0..127 tile row
    const int ldc = (tid & 1) << 4;    // 0 or 16

    const unsigned short* gA = A  + (bm + ldr) * (size_t)K + ldc;
    const unsigned short* gB = Bt + (bn + ldr) * (size_t)K + ldc;

    // low 32 bits of a shared-memory address == LDS byte offset (AS3)
    unsigned ldsA[2], ldsB[2];
    ldsA[0] = (unsigned)(size_t)&sA[0][ldr * LDSP + ldc];
    ldsA[1] = (unsigned)(size_t)&sA[1][ldr * LDSP + ldc];
    ldsB[0] = (unsigned)(size_t)&sB[0][ldr * LDSP + ldc];
    ldsB[1] = (unsigned)(size_t)&sB[1][ldr * LDSP + ldc];

    v8f acc[2][4];
#pragma unroll
    for (int i = 0; i < 2; ++i)
#pragma unroll
        for (int j = 0; j < 4; ++j) acc[i][j] = v8f{};

    // prologue: stage k-tile 0 into buffer 0 (4 async b128 per thread-pair set)
    ASYNC_CP32(ldsA[0], gA);
    ASYNC_CP32(ldsB[0], gB);

    const int nk = K / BK;
    for (int it = 0; it < nk; ++it) {
        const int buf = it & 1;
        if (it + 1 < nk) {
            // stream next tile into the other buffer while we compute
            const unsigned short* nA = gA + (size_t)(it + 1) * BK;
            const unsigned short* nB = gB + (size_t)(it + 1) * BK;
            ASYNC_CP32(ldsA[buf ^ 1], nA);
            ASYNC_CP32(ldsB[buf ^ 1], nB);
            // 4 newly-issued copies outstanding; <=4 => current tile landed
            asm volatile("s_wait_asynccnt 0x4" ::: "memory");
        } else {
            asm volatile("s_wait_asynccnt 0x0" ::: "memory");
        }
        __syncthreads();   // all waves' async copies for `buf` visible

        // A fragment (16x32, 16-bit): lane(l<16)=row M=l: v0..3 = K[hi*8..+7],
        // v4..7 = K[16+hi*8..+7]  (hi = lane/16)
        FragAB fa[2], fb[4];
#pragma unroll
        for (int fm = 0; fm < 2; ++fm) {
            int m = wm * 32 + fm * 16 + l16;
            fa[fm].h[0] = *(const v8bf*)(&sA[buf][m * LDSP + half * 8]);
            fa[fm].h[1] = *(const v8bf*)(&sA[buf][m * LDSP + 16 + half * 8]);
        }
        // B fragment (32x16, 16-bit): lane n holds col n; v0..3 = K[hi*8..+7], v4..7 = K[16+hi*8..+7]
#pragma unroll
        for (int fn = 0; fn < 4; ++fn) {
            int n = wn * 64 + fn * 16 + l16;
            fb[fn].h[0] = *(const v8bf*)(&sB[buf][n * LDSP + half * 8]);
            fb[fn].h[1] = *(const v8bf*)(&sB[buf][n * LDSP + 16 + half * 8]);
        }
#pragma unroll
        for (int fm = 0; fm < 2; ++fm)
#pragma unroll
            for (int fn = 0; fn < 4; ++fn)
                acc[fm][fn] = __builtin_amdgcn_wmma_f32_16x16x32_bf16(
                    false, fa[fm].v, false, fb[fn].v,
                    (short)0, acc[fm][fn], false, false);

        __syncthreads();   // fragment reads done before buffer is overwritten
    }

    // Epilogue. C/D layout: VGPR r -> row r + 8*(lane/16), col = lane%16.
#pragma unroll
    for (int fm = 0; fm < 2; ++fm) {
#pragma unroll
        for (int fn = 0; fn < 4; ++fn) {
            int col = (int)bn + wn * 64 + fn * 16 + l16;
            float bv = bias[col];
            size_t rbase = bm + (size_t)(wm * 32 + fm * 16 + half * 8);
#pragma unroll
            for (int r = 0; r < 8; ++r) {
                size_t o = (rbase + r) * (size_t)Nn + col;
                float v = acc[fm][fn][r] + bv;
                if (mode == 0)      outF[o] = v;
                else if (mode == 1) outH[o] = f2bf(v);
                else                outF[o] += v;
            }
        }
    }
}

// ---------------------------------------------------------------------------
// Transpose + convert weights: W[K x N] f32  ->  Wt[N x K] bf16
// ---------------------------------------------------------------------------
__global__ void transpose_to_bf16(const float* __restrict__ W,
                                  unsigned short* __restrict__ Wt,
                                  int K, int N)
{
    int idx = blockIdx.x * blockDim.x + threadIdx.x;
    if (idx >= K * N) return;
    int k = idx / N, n = idx - k * N;
    Wt[(size_t)n * K + k] = f2bf(W[idx]);
}

// f32 -> bf16 (optional relu)
__global__ void convert_bf16_kernel(const float* __restrict__ src,
                                    unsigned short* __restrict__ dst,
                                    int n, int do_relu)
{
    int i = blockIdx.x * blockDim.x + threadIdx.x;
    if (i >= n) return;
    float v = src[i];
    if (do_relu) v = fmaxf(v, 0.0f);
    dst[i] = f2bf(v);
}

// ---------------------------------------------------------------------------
// Per-channel online softmax partials over node axis.
// m = relu(x) + 1e-7 ; p = t*m ; track (max, S=sum e^{p-max}, T=sum m e^{p-max})
// block = 512 threads (one channel each), rows chunked per block.
// ---------------------------------------------------------------------------
__global__ __launch_bounds__(512) void softmax_reduce1(
    const float* __restrict__ x, const float* __restrict__ tptr,
    float* __restrict__ red, int rows_per_block)
{
    int c = threadIdx.x;                 // channel 0..511
    float t = *tptr;
    int r0 = blockIdx.x * rows_per_block;
    float mx = -3.4028235e38f, S = 0.f, T = 0.f;
    for (int r = r0; r < r0 + rows_per_block; ++r) {
        float m = fmaxf(x[(size_t)r * 512 + c], 0.0f) + 1e-7f;
        float p = t * m;
        if (p > mx) {
            float sc = __expf(mx - p);
            S *= sc; T *= sc; mx = p;
        }
        float e = __expf(p - mx);
        S += e; T += m * e;
    }
    float* o = red + ((size_t)blockIdx.x * 512 + c) * 3;
    o[0] = mx; o[1] = S; o[2] = T;
}

__global__ __launch_bounds__(512) void softmax_reduce2(
    const float* __restrict__ red, float* __restrict__ agg, int nblocks)
{
    int c = threadIdx.x;
    float mx = -3.4028235e38f, S = 0.f, T = 0.f;
    for (int b = 0; b < nblocks; ++b) {
        const float* p = red + ((size_t)b * 512 + c) * 3;
        float m = p[0], s = p[1], tt = p[2];
        float nm = fmaxf(mx, m);
        float e0 = __expf(mx - nm), e1 = __expf(m - nm);
        S = S * e0 + s * e1;
        T = T * e0 + tt * e1;
        mx = nm;
    }
    agg[c] = T / S;
}

// bufA = bf16( relu(x) + agg[channel] )   (x width fixed at 512)
__global__ void prep_add_agg(const float* __restrict__ x,
                             const float* __restrict__ agg,
                             unsigned short* __restrict__ dst, int total)
{
    int i = blockIdx.x * blockDim.x + threadIdx.x;
    if (i >= total) return;
    int c = i & 511;
    dst[i] = f2bf(fmaxf(x[i], 0.0f) + agg[c]);
}

// ---------------------------------------------------------------------------
// In-place LayerNorm (over width=1024) + ReLU on bf16 buffer, f32 statistics.
// One block (256 threads) per row, 4 elems/thread.
// ---------------------------------------------------------------------------
__global__ __launch_bounds__(256) void ln_relu_kernel(
    unsigned short* __restrict__ z,
    const float* __restrict__ g, const float* __restrict__ be)
{
    __shared__ float sh[256];
    const int W = 1024;
    unsigned short* zr = z + (size_t)blockIdx.x * W;
    int base = threadIdx.x * 4;

    float v[4], s = 0.f, s2 = 0.f;
#pragma unroll
    for (int i = 0; i < 4; ++i) {
        float f = bf2f(zr[base + i]);
        v[i] = f; s += f; s2 += f * f;
    }
    sh[threadIdx.x] = s;  __syncthreads();
    for (int o = 128; o > 0; o >>= 1) {
        if (threadIdx.x < o) sh[threadIdx.x] += sh[threadIdx.x + o];
        __syncthreads();
    }
    float stot = sh[0]; __syncthreads();
    sh[threadIdx.x] = s2; __syncthreads();
    for (int o = 128; o > 0; o >>= 1) {
        if (threadIdx.x < o) sh[threadIdx.x] += sh[threadIdx.x + o];
        __syncthreads();
    }
    float s2tot = sh[0];

    float mu  = stot * (1.0f / W);
    float var = s2tot * (1.0f / W) - mu * mu;
    float inv = rsqrtf(var + 1e-5f);
#pragma unroll
    for (int i = 0; i < 4; ++i) {
        int c = base + i;
        float y = (v[i] - mu) * inv * g[c] + be[c];
        zr[c] = f2bf(fmaxf(y, 0.0f));
    }
}

// ---------------------------------------------------------------------------
// Host-side orchestration
// ---------------------------------------------------------------------------
extern "C" void kernel_launch(void* const* d_in, const int* in_sizes, int n_in,
                              void* d_out, int out_size, void* d_ws, size_t ws_size,
                              hipStream_t stream)
{
    (void)in_sizes; (void)n_in; (void)out_size; (void)ws_size;
    const int N = 32768, F_IN = 64, D = 512, D2 = 1024, D_OUT = 128;

    const float* batch = (const float*)d_in[0];
    const float* W_enc = (const float*)d_in[1];
    const float* b_enc = (const float*)d_in[2];
    const float* Wf    = (const float*)d_in[3];
    const float* bfv   = (const float*)d_in[4];
    const float* t_p [2] = { (const float*)d_in[5],  (const float*)d_in[12] };
    const float* W1_p[2] = { (const float*)d_in[6],  (const float*)d_in[13] };
    const float* b1_p[2] = { (const float*)d_in[7],  (const float*)d_in[14] };
    const float* g_p [2] = { (const float*)d_in[8],  (const float*)d_in[15] };
    const float* be_p[2] = { (const float*)d_in[9],  (const float*)d_in[16] };
    const float* W2_p[2] = { (const float*)d_in[10], (const float*)d_in[17] };
    const float* b2_p[2] = { (const float*)d_in[11], (const float*)d_in[18] };
    float* out = (float*)d_out;

    // workspace carve-up (256B aligned)
    char* ws = (char*)d_ws;
    size_t off = 0;
    auto carve = [&](size_t bytes) -> char* {
        char* p = ws + off;
        off = (off + bytes + 255) & ~(size_t)255;
        return p;
    };
    float*          x    = (float*)         carve((size_t)N * D * 4);       // 64 MB
    unsigned short* bufZ = (unsigned short*)carve((size_t)N * D2 * 2);      // 64 MB
    unsigned short* bufA = (unsigned short*)carve((size_t)N * D * 2);       // 32 MB
    unsigned short* WtE  = (unsigned short*)carve((size_t)D * F_IN * 2);
    unsigned short* WtF  = (unsigned short*)carve((size_t)D_OUT * D * 2);
    unsigned short* Wt1[2], *Wt2[2];
    for (int l = 0; l < 2; ++l) {
        Wt1[l] = (unsigned short*)carve((size_t)D2 * D * 2);
        Wt2[l] = (unsigned short*)carve((size_t)D * D2 * 2);
    }
    const int RED_BLOCKS = 128;
    float* red = (float*)carve((size_t)RED_BLOCKS * 512 * 3 * 4);
    float* agg = (float*)carve(512 * 4);

    auto cdiv = [](int a, int b) { return (a + b - 1) / b; };

    // 1) weights -> transposed bf16
    transpose_to_bf16<<<cdiv(F_IN * D, 256), 256, 0, stream>>>(W_enc, WtE, F_IN, D);
    transpose_to_bf16<<<cdiv(D * D_OUT, 256), 256, 0, stream>>>(Wf, WtF, D, D_OUT);
    for (int l = 0; l < 2; ++l) {
        transpose_to_bf16<<<cdiv(D * D2, 256), 256, 0, stream>>>(W1_p[l], Wt1[l], D, D2);
        transpose_to_bf16<<<cdiv(D2 * D, 256), 256, 0, stream>>>(W2_p[l], Wt2[l], D2, D);
    }

    // 2) encoder: x = batch @ W_enc + b_enc
    convert_bf16_kernel<<<cdiv(N * F_IN, 256), 256, 0, stream>>>(batch, bufA, N * F_IN, 0);
    gemm_bf16_wmma<<<dim3(N / BM, D / BN), 256, 0, stream>>>(
        bufA, WtE, b_enc, x, nullptr, D, F_IN, /*mode=*/0);

    // 3) two DeepGCN layers
    for (int l = 0; l < 2; ++l) {
        softmax_reduce1<<<RED_BLOCKS, 512, 0, stream>>>(x, t_p[l], red, N / RED_BLOCKS);
        softmax_reduce2<<<1, 512, 0, stream>>>(red, agg, RED_BLOCKS);
        prep_add_agg<<<cdiv(N * D, 256), 256, 0, stream>>>(x, agg, bufA, N * D);
        // z = (relu(x)+agg) @ W1 + b1   (bf16 out)
        gemm_bf16_wmma<<<dim3(N / BM, D2 / BN), 256, 0, stream>>>(
            bufA, Wt1[l], b1_p[l], nullptr, bufZ, D2, D, /*mode=*/1);
        // z = relu(LN(z))  in place
        ln_relu_kernel<<<N, 256, 0, stream>>>(bufZ, g_p[l], be_p[l]);
        // x += z @ W2 + b2
        gemm_bf16_wmma<<<dim3(N / BM, D / BN), 256, 0, stream>>>(
            bufZ, Wt2[l], b2_p[l], x, nullptr, D, D2, /*mode=*/2);
    }

    // 4) out = relu(x) @ Wf + bf
    convert_bf16_kernel<<<cdiv(N * D, 256), 256, 0, stream>>>(x, bufA, N * D, 1);
    gemm_bf16_wmma<<<dim3(N / BM, D_OUT / BN), 256, 0, stream>>>(
        bufA, WtF, bfv, out, nullptr, D_OUT, D, /*mode=*/0);
}
